// GATReg_add_70592082477426
// MI455X (gfx1250) — matrix operand order repaired and checked by reading, not compiled, not run on hardware
//
#include <hip/hip_runtime.h>

// ---------------- problem constants (from reference) ----------------
#define NN      50000      // nodes
#define EE      800000     // edges
#define GG      500        // graphs
#define INDIM   128
#define EXTRA   32
#define HID     64
#define HEADS   4
#define F2      256        // HEADS*HID
#define FINAL   288        // F2 + EXTRA
#define NPG     100        // nodes per graph

typedef _Float16 v16h __attribute__((ext_vector_type(16)));
typedef _Float16 v8h  __attribute__((ext_vector_type(8)));
typedef float    v8f  __attribute__((ext_vector_type(8)));

// ---------------- helpers ----------------
__device__ __forceinline__ unsigned f2ord(float f) {
    unsigned u = __float_as_uint(f);
    return (u & 0x80000000u) ? ~u : (u | 0x80000000u);
}
__device__ __forceinline__ float ord2f(unsigned o) {
    unsigned u = (o & 0x80000000u) ? (o & 0x7FFFFFFFu) : ~o;
    return __uint_as_float(u);
}

// ---------------- f32 -> f16 convert ----------------
__global__ void k_cvt_f16(const float* __restrict__ in, _Float16* __restrict__ out, int n) {
    int i = blockIdx.x * blockDim.x + threadIdx.x;
    if (i < n) out[i] = (_Float16)in[i];
}

// ---------------- pack row-major f32 W[K,N] into WMMA B-operand layout (f16) --------
// packed[p]: tile = p>>9 (512 halves per 32x16 K-tile x N-tile), lane = (p>>4)&31, j = p&15
// lane holds col n = nt*16 + (lane&15); half j holds K = ktile*32 + 16*(lane>>4) + j
__global__ void k_pack_b(const float* __restrict__ W, _Float16* __restrict__ P, int K, int Nn) {
    int p = blockIdx.x * blockDim.x + threadIdx.x;
    int total = K * Nn;
    if (p >= total) return;
    int tile   = p >> 9;
    int within = p & 511;
    int lane   = within >> 4;
    int j      = within & 15;
    int ntiles = Nn >> 4;
    int ktile  = tile / ntiles;
    int nt     = tile % ntiles;
    int k = ktile * 32 + ((lane >> 4) << 4) + j;
    int n = (nt << 4) + (lane & 15);
    P[p] = (_Float16)W[(size_t)k * Nn + n];
}

// ---------------- WMMA GEMM: C[M,Nn] = A[M,K](f16,row-major) * Bpacked ----------------
// one wave computes a 16x64 output tile; 4 v8f accumulators; K in steps of 32
__global__ void __launch_bounds__(256) k_gemm_wmma(const _Float16* __restrict__ A,
                                                   const _Float16* __restrict__ Bp,
                                                   float* __restrict__ C,
                                                   int M, int K, int Nn) {
    int wave     = blockIdx.x * (blockDim.x >> 5) + (threadIdx.x >> 5);
    int colTiles = Nn >> 6;                 // 64-wide tiles
    int total    = (M >> 4) * colTiles;
    if (wave >= total) return;              // wave-uniform guard: EXEC stays all-1s
    int rowTile = wave / colTiles;
    int colTile = wave % colTiles;
    int lane = threadIdx.x & 31;
    int half = lane >> 4;
    int lm   = lane & 15;
    int row  = (rowTile << 4) + lm;
    int nTilesTotal = Nn >> 4;

    v8f acc0 = {}, acc1 = {}, acc2 = {}, acc3 = {};
    for (int k0 = 0; k0 < K; k0 += 32) {
        // A operand: lane row = lm; halves 0..15 = K k0+16*half..
        const _Float16* arow = A + (size_t)row * K + k0 + (half << 4);
        v16h a = *(const v16h*)(arow);

        const _Float16* bbase =
            Bp + ((size_t)(k0 >> 5) * nTilesTotal + ((size_t)colTile << 2)) * 512 + (lane << 4);
        v16h b0 = *(const v16h*)(bbase);
        v16h b1 = *(const v16h*)(bbase + 512);
        v16h b2 = *(const v16h*)(bbase + 1024);
        v16h b3 = *(const v16h*)(bbase + 1536);

        acc0 = __builtin_amdgcn_wmma_f32_16x16x32_f16(false, a, false, b0, (short)0, acc0, false, false);
        acc1 = __builtin_amdgcn_wmma_f32_16x16x32_f16(false, a, false, b1, (short)0, acc1, false, false);
        acc2 = __builtin_amdgcn_wmma_f32_16x16x32_f16(false, a, false, b2, (short)0, acc2, false, false);
        acc3 = __builtin_amdgcn_wmma_f32_16x16x32_f16(false, a, false, b3, (short)0, acc3, false, false);
    }
    // D layout: lane holds col = lm, vgpr r holds row r + 8*half
    int colBase = colTile << 6;
    float* crow = C + (size_t)((rowTile << 4) + (half << 3)) * Nn + colBase + lm;
#pragma unroll
    for (int r = 0; r < 8; ++r) {
        crow[(size_t)r * Nn + 0]  = acc0[r];
        crow[(size_t)r * Nn + 16] = acc1[r];
        crow[(size_t)r * Nn + 32] = acc2[r];
        crow[(size_t)r * Nn + 48] = acc3[r];
    }
}

// ---------------- per-node attention logits el/er ----------------
__global__ void k_attn_lr(const float* __restrict__ feat, const float* __restrict__ al,
                          const float* __restrict__ ar, float* __restrict__ el,
                          float* __restrict__ er) {
    int i = blockIdx.x * blockDim.x + threadIdx.x;  // n*HEADS + h
    if (i >= NN * HEADS) return;
    int h = i & 3;
    int n = i >> 2;
    const float* f  = feat + (size_t)n * F2 + h * HID;
    const float* wl = al + h * HID;
    const float* wr = ar + h * HID;
    float sl = 0.f, sr = 0.f;
#pragma unroll 16
    for (int d = 0; d < HID; ++d) { float v = f[d]; sl += v * wl[d]; sr += v * wr[d]; }
    el[i] = sl;
    er[i] = sr;
}

// ---------------- edge: leaky-relu logit + segment max (ordered-uint atomicMax) ------
__global__ void k_edge_max(const int* __restrict__ src, const int* __restrict__ dst,
                           const float* __restrict__ el, const float* __restrict__ er,
                           float* __restrict__ ebuf, unsigned* __restrict__ mord) {
    int i = blockIdx.x * blockDim.x + threadIdx.x;
    if (i >= EE * HEADS) return;
    int h = i & 3;
    int e = i >> 2;
    int s = src[e], d = dst[e];
    float v = el[s * HEADS + h] + er[d * HEADS + h];
    v = v > 0.f ? v : 0.2f * v;
    ebuf[i] = v;
    atomicMax(&mord[d * HEADS + h], f2ord(v));
}

// ---------------- edge: exp(e - m[dst]) + segment sum denom ----------------
__global__ void k_edge_exp(const int* __restrict__ dst, float* __restrict__ ebuf,
                           const unsigned* __restrict__ mord, float* __restrict__ denom) {
    int i = blockIdx.x * blockDim.x + threadIdx.x;
    if (i >= EE * HEADS) return;
    int h = i & 3;
    int e = i >> 2;
    int d = dst[e];
    float m  = ord2f(mord[d * HEADS + h]);
    float ex = __expf(ebuf[i] - m);
    ebuf[i] = ex;
    atomicAdd(&denom[d * HEADS + h], ex);
}

// ---------------- edge: alpha-weighted message scatter (4 channels/thread) ----------
__global__ void k_scatter(const int* __restrict__ src, const int* __restrict__ dst,
                          const float* __restrict__ feat, const float* __restrict__ ebuf,
                          const float* __restrict__ denom, float* __restrict__ rst) {
    long long i = (long long)blockIdx.x * blockDim.x + threadIdx.x;  // over E*64
    if (i >= (long long)EE * 64) return;
    int e  = (int)(i >> 6);
    int q  = (int)(i & 63);
    int c0 = q << 2;
    int h  = q >> 4;
    int s = src[e], d = dst[e];
    float alpha = ebuf[e * HEADS + h] / denom[d * HEADS + h];
    const float4 f4 = *(const float4*)(feat + (size_t)s * F2 + c0);
    float* out = rst + (size_t)d * F2 + c0;
    atomicAdd(out + 0, f4.x * alpha);
    atomicAdd(out + 1, f4.y * alpha);
    atomicAdd(out + 2, f4.z * alpha);
    atomicAdd(out + 3, f4.w * alpha);
}

// ---------------- layer-1 epilogue: relu(rst + b) -> f16 (GEMM-2 input) -------------
__global__ void k_relu_bias_h16(const float* __restrict__ rst, const float* __restrict__ b,
                                _Float16* __restrict__ out, int total) {
    int i = blockIdx.x * blockDim.x + threadIdx.x;
    if (i >= total) return;
    int c = i & (F2 - 1);
    float v = rst[i] + b[c];
    out[i] = (_Float16)(v > 0.f ? v : 0.f);
}

// ---------------- layer-2 epilogue fused into per-graph mean pooling ----------------
__global__ void __launch_bounds__(256) k_pool(const float* __restrict__ rst,
                                              const float* __restrict__ b2,
                                              float* __restrict__ hg) {
    int g = blockIdx.x;
    int c = threadIdx.x;           // 256 channels
    float bias = b2[c];
    float s = 0.f;
    const float* base = rst + (size_t)g * NPG * F2 + c;
    for (int i = 0; i < NPG; ++i) {
        float v = base[(size_t)i * F2] + bias;
        s += v > 0.f ? v : 0.f;
    }
    hg[(size_t)g * F2 + c] = s * (1.0f / NPG);
}

// ---------------- MLP head: one 64-thread block per graph ----------------
__global__ void __launch_bounds__(64) k_head(const float* __restrict__ hg,
                                             const float* __restrict__ desc,
                                             const float* __restrict__ fc1w, const float* __restrict__ fc1b,
                                             const float* __restrict__ fc2w, const float* __restrict__ fc2b,
                                             const float* __restrict__ outw, const float* __restrict__ outb,
                                             float* __restrict__ out) {
    __shared__ float comb[FINAL];
    __shared__ float z1[64];
    __shared__ float z2[32];
    int g = blockIdx.x;
    int t = threadIdx.x;
    for (int i = t; i < F2; i += 64) comb[i] = hg[(size_t)g * F2 + i];
    if (t < EXTRA) comb[F2 + t] = desc[(size_t)g * EXTRA + t];
    __syncthreads();
    float s = fc1b[t];
    for (int i = 0; i < FINAL; ++i) s += comb[i] * fc1w[i * 64 + t];
    z1[t] = s > 0.f ? s : 0.f;
    __syncthreads();
    if (t < 32) {
        float s2 = fc2b[t];
        for (int i = 0; i < 64; ++i) s2 += z1[i] * fc2w[i * 32 + t];
        z2[t] = s2 > 0.f ? s2 : 0.f;
    }
    __syncthreads();
    if (t == 0) {
        float s3 = outb[0];
        for (int i = 0; i < 32; ++i) s3 += z2[i] * outw[i];
        out[g] = s3;
    }
}

// ---------------- host orchestration ----------------
extern "C" void kernel_launch(void* const* d_in, const int* in_sizes, int n_in,
                              void* d_out, int out_size, void* d_ws, size_t ws_size,
                              hipStream_t stream) {
    const float* x    = (const float*)d_in[0];
    const float* desc = (const float*)d_in[1];
    const int*   src  = (const int*)d_in[2];
    const int*   dst  = (const int*)d_in[3];
    // d_in[4] graph_id unused: graphs are contiguous blocks of NPG nodes
    const float* W1   = (const float*)d_in[5];
    const float* al1  = (const float*)d_in[6];
    const float* ar1  = (const float*)d_in[7];
    const float* b1   = (const float*)d_in[8];
    const float* W2   = (const float*)d_in[9];
    const float* al2  = (const float*)d_in[10];
    const float* ar2  = (const float*)d_in[11];
    const float* b2   = (const float*)d_in[12];
    const float* fc1w = (const float*)d_in[13];
    const float* fc1b = (const float*)d_in[14];
    const float* fc2w = (const float*)d_in[15];
    const float* fc2b = (const float*)d_in[16];
    const float* outw = (const float*)d_in[17];
    const float* outb = (const float*)d_in[18];
    float* out = (float*)d_out;

    // workspace carve-out (256B aligned slices)
    char* ws = (char*)d_ws;
    auto alloc = [&](size_t bytes) -> char* {
        char* p = ws;
        ws += (bytes + 255) & ~(size_t)255;
        return p;
    };
    float*    feat  = (float*)   alloc((size_t)NN * F2 * 4);   // GEMM output (both layers)
    float*    rst   = (float*)   alloc((size_t)NN * F2 * 4);   // scatter accumulator
    _Float16* xh    = (_Float16*)alloc((size_t)NN * INDIM * 2);
    _Float16* h1h   = (_Float16*)alloc((size_t)NN * F2 * 2);
    _Float16* W1p   = (_Float16*)alloc((size_t)INDIM * F2 * 2);
    _Float16* W2p   = (_Float16*)alloc((size_t)F2 * F2 * 2);
    float*    el    = (float*)   alloc((size_t)NN * HEADS * 4);
    float*    er    = (float*)   alloc((size_t)NN * HEADS * 4);
    unsigned* mord  = (unsigned*)alloc((size_t)NN * HEADS * 4);
    float*    denom = (float*)   alloc((size_t)NN * HEADS * 4);
    float*    ebuf  = (float*)   alloc((size_t)EE * HEADS * 4);
    float*    hg    = (float*)   alloc((size_t)GG * F2 * 4);

    const int B = 256;
    int nEH   = EE * HEADS;
    int gEH   = (nEH + B - 1) / B;
    int gNH   = (NN * HEADS + B - 1) / B;
    int gNF   = (NN * F2 + B - 1) / B;
    long long scatN = (long long)EE * 64;
    int gScat = (int)((scatN + B - 1) / B);
    int gemmWaves = (NN / 16) * (F2 / 64);
    int gGemm = (gemmWaves + 7) / 8;

    // --- prep: converts + weight packs ---
    k_cvt_f16<<<(NN * INDIM + B - 1) / B, B, 0, stream>>>(x, xh, NN * INDIM);
    k_pack_b <<<(INDIM * F2 + B - 1) / B, B, 0, stream>>>(W1, W1p, INDIM, F2);
    k_pack_b <<<(F2 * F2 + B - 1) / B, B, 0, stream>>>(W2, W2p, F2, F2);

    // ================= layer 1 =================
    k_gemm_wmma<<<gGemm, B, 0, stream>>>(xh, W1p, feat, NN, INDIM, F2);
    k_attn_lr<<<gNH, B, 0, stream>>>(feat, al1, ar1, el, er);
    hipMemsetAsync(mord, 0, (size_t)NN * HEADS * 4, stream);   // ordered(-max) == 0
    hipMemsetAsync(denom, 0, (size_t)NN * HEADS * 4, stream);
    hipMemsetAsync(rst, 0, (size_t)NN * F2 * 4, stream);
    k_edge_max<<<gEH, B, 0, stream>>>(src, dst, el, er, ebuf, mord);
    k_edge_exp<<<gEH, B, 0, stream>>>(dst, ebuf, mord, denom);
    k_scatter<<<gScat, B, 0, stream>>>(src, dst, feat, ebuf, denom, rst);
    k_relu_bias_h16<<<gNF, B, 0, stream>>>(rst, b1, h1h, NN * F2);

    // ================= layer 2 =================
    k_gemm_wmma<<<gGemm, B, 0, stream>>>(h1h, W2p, feat, NN, F2, F2);
    k_attn_lr<<<gNH, B, 0, stream>>>(feat, al2, ar2, el, er);
    hipMemsetAsync(mord, 0, (size_t)NN * HEADS * 4, stream);
    hipMemsetAsync(denom, 0, (size_t)NN * HEADS * 4, stream);
    hipMemsetAsync(rst, 0, (size_t)NN * F2 * 4, stream);
    k_edge_max<<<gEH, B, 0, stream>>>(src, dst, el, er, ebuf, mord);
    k_edge_exp<<<gEH, B, 0, stream>>>(dst, ebuf, mord, denom);
    k_scatter<<<gScat, B, 0, stream>>>(src, dst, feat, ebuf, denom, rst);

    // ================= pooling + head =================
    k_pool<<<GG, 256, 0, stream>>>(rst, b2, hg);
    k_head<<<GG, 64, 0, stream>>>(hg, desc, fc1w, fc1b, fc2w, fc2b, outw, outb, out);
}
